// CausalSelfAttention_53042846105638
// MI455X (gfx1250) — compile-verified, matrix operand order checked by reading, not run on hardware
//
#include <hip/hip_runtime.h>

// ---------------------------------------------------------------------------
// CDNA5 (gfx1250) causal self-attention, all matmuls via v_wmma_f32_16x16x32_f16.
// B=2, T=4096, d_model=768, H=12, Dh=64.
// GEMMs: 32x64 tile/wave, double-buffered K-loop (overlap loads with WMMA).
// Attention: flash-style, 16 q-rows/wave, keys in tiles of 32, V loads hoisted.
// ---------------------------------------------------------------------------

typedef __attribute__((ext_vector_type(16))) _Float16 v16h;
typedef __attribute__((ext_vector_type(8)))  _Float16 h8;
typedef __attribute__((ext_vector_type(8)))  float    v8f;

union AFrag { v16h v; h8 h[2]; };

__device__ __forceinline__ h8 load_h8(const _Float16* p) {
    return *reinterpret_cast<const h8*>(p);
}

__device__ __forceinline__ v8f wmma16x16x32(const AFrag& a, const AFrag& b, v8f c) {
    // D = A(16x32,f16) x B(32x16,f16) + C(16x16,f32)
    return __builtin_amdgcn_wmma_f32_16x16x32_f16(
        /*neg_a=*/false, a.v, /*neg_b=*/false, b.v,
        /*c_mod=*/(short)0, c, /*reuse_a=*/false, /*reuse_b=*/false);
}

#define BATCH   2
#define SEQ     4096
#define DMODEL  768
#define NHEADS  12
#define DHEAD   64
#define ROWS    (BATCH * SEQ)          // 8192

// ---------------------------------------------------------------------------
// f32 -> f16 convert
// ---------------------------------------------------------------------------
__global__ void convert_f32_f16(const float* __restrict__ in,
                                _Float16* __restrict__ out, int n) {
    int i = blockIdx.x * blockDim.x + threadIdx.x;
    if (i < n) out[i] = (_Float16)in[i];
}

// ---------------------------------------------------------------------------
// QKV projection: qkv[row][e] = sum_k x[row][k] * w_qkv[e][k]
// One wave computes a 32x64 tile; K-loop double-buffered.
// Writes Q (scaled by 1/8) and K as [B,H,T,64] f16, V transposed [B,H,64,T].
// ---------------------------------------------------------------------------
__global__ __launch_bounds__(128) void qkv_gemm(
    const _Float16* __restrict__ xb, const _Float16* __restrict__ wq,
    _Float16* __restrict__ Qh, _Float16* __restrict__ Kh,
    _Float16* __restrict__ Vt)
{
    const int wave = threadIdx.x >> 5;
    const int lane = threadIdx.x & 31;
    const int task = blockIdx.x * 4 + wave;   // 256 row-tiles(32) * 36 col-tiles(64)
    const int rt = task / 36;
    const int ct = task % 36;
    const int m  = lane & 15;                 // A row / B col / C col
    const int kh = lane >> 4;                 // K-half select

    const _Float16* xrow0 = xb + (size_t)(rt * 32 + m) * DMODEL;
    const _Float16* xrow1 = xrow0 + (size_t)16 * DMODEL;
    const _Float16* wbase = wq + (size_t)(ct * 64 + m) * DMODEL;

    AFrag A[2][2], B[2][4];
    v8f acc[8] = {};

    auto loadA = [&](AFrag* dst, int kt) {
        dst[0].h[0] = load_h8(xrow0 + kt + kh * 8);
        dst[0].h[1] = load_h8(xrow0 + kt + 16 + kh * 8);
        dst[1].h[0] = load_h8(xrow1 + kt + kh * 8);
        dst[1].h[1] = load_h8(xrow1 + kt + 16 + kh * 8);
    };
    auto loadB = [&](AFrag* dst, int kt) {
        #pragma unroll
        for (int sub = 0; sub < 4; ++sub) {
            const _Float16* wrow = wbase + (size_t)sub * 16 * DMODEL + kt;
            dst[sub].h[0] = load_h8(wrow + kh * 8);
            dst[sub].h[1] = load_h8(wrow + 16 + kh * 8);
        }
    };

    loadA(A[0], 0);
    loadB(B[0], 0);
    int pb = 0;
    for (int kt = 0; kt < DMODEL; kt += 32, pb ^= 1) {
        if (kt + 32 < DMODEL) {               // prefetch next K-slice
            loadA(A[pb ^ 1], kt + 32);
            loadB(B[pb ^ 1], kt + 32);
        }
        #pragma unroll
        for (int sub = 0; sub < 4; ++sub)
            acc[sub] = wmma16x16x32(A[pb][0], B[pb][sub], acc[sub]);
        #pragma unroll
        for (int sub = 0; sub < 4; ++sub)
            acc[4 + sub] = wmma16x16x32(A[pb][1], B[pb][sub], acc[4 + sub]);
    }

    #pragma unroll
    for (int rb = 0; rb < 2; ++rb) {
        #pragma unroll
        for (int sub = 0; sub < 4; ++sub) {
            const int e   = ct * 64 + sub * 16 + m;  // out-feature in [0,2304)
            const int sel = e / DMODEL;              // 0=Q 1=K 2=V
            const int d   = e % DMODEL;
            const int h   = d >> 6;
            const int dh  = d & 63;
            #pragma unroll
            for (int v = 0; v < 8; ++v) {
                const int row = rt * 32 + rb * 16 + v + 8 * kh;  // C row = v + 8*hi
                const int bi  = row >> 12;
                const int t   = row & (SEQ - 1);
                const float val = acc[rb * 4 + sub][v];
                if (sel == 0)
                    Qh[(((size_t)bi * NHEADS + h) * SEQ + t) * DHEAD + dh] =
                        (_Float16)(val * 0.125f);    // fold 1/sqrt(64)
                else if (sel == 1)
                    Kh[(((size_t)bi * NHEADS + h) * SEQ + t) * DHEAD + dh] = (_Float16)val;
                else
                    Vt[(((size_t)bi * NHEADS + h) * DHEAD + dh) * SEQ + t] = (_Float16)val;
            }
        }
    }
}

// ---------------------------------------------------------------------------
// Flash attention: one wave per (b,h,16-row q tile); keys in tiles of 32.
// Online softmax in C-layout (row = v + 8*hi, col = lane&15); P transposed
// C-layout -> A-layout through LDS (in-order DS within the single wave).
// V-fragment loads hoisted next to K loads so they overlap softmax VALU work.
// ---------------------------------------------------------------------------
__global__ __launch_bounds__(32) void attn_kernel(
    const _Float16* __restrict__ Qh, const _Float16* __restrict__ Kh,
    const _Float16* __restrict__ Vt, _Float16* __restrict__ attnb)
{
    __shared__ __align__(16) _Float16 pt[16][40];   // 16x32 P tile, padded rows

    const int lane = threadIdx.x & 31;
    const int n    = lane & 15;
    const int kh   = lane >> 4;
    const int qt   = blockIdx.x & 255;
    const int bh   = blockIdx.x >> 8;               // b*12 + h
    const int q0   = qt * 16;

    const _Float16* Qp = Qh + (size_t)bh * SEQ * DHEAD;
    const _Float16* Kp = Kh + (size_t)bh * SEQ * DHEAD;
    const _Float16* Vp = Vt + (size_t)bh * DHEAD * SEQ;

    // Q A-fragments for d=0..31 and d=32..63
    AFrag aq0, aq1;
    {
        const _Float16* qrow = Qp + (size_t)(q0 + n) * DHEAD;
        aq0.h[0] = load_h8(qrow + kh * 8);
        aq0.h[1] = load_h8(qrow + 16 + kh * 8);
        aq1.h[0] = load_h8(qrow + 32 + kh * 8);
        aq1.h[1] = load_h8(qrow + 48 + kh * 8);
    }

    v8f o[4] = {};
    float mrun[8], lrun[8];
    #pragma unroll
    for (int v = 0; v < 8; ++v) { mrun[v] = -__builtin_inff(); lrun[v] = 0.f; }

    const int jmax = (q0 + 47) >> 5;   // key tiles of 32 covering keys 0..q0+15
    for (int j = 0; j < jmax; ++j) {
        const int kbase = j * 32;

        // --- issue ALL loads for this tile up front (K frags, then V frags) ---
        AFrag b00, b01, b10, b11, bv[4];
        {
            const _Float16* k0 = Kp + (size_t)(kbase + n) * DHEAD;
            b00.h[0] = load_h8(k0 + kh * 8);      b00.h[1] = load_h8(k0 + 16 + kh * 8);
            b01.h[0] = load_h8(k0 + 32 + kh * 8); b01.h[1] = load_h8(k0 + 48 + kh * 8);
            const _Float16* k1 = k0 + 16 * DHEAD;
            b10.h[0] = load_h8(k1 + kh * 8);      b10.h[1] = load_h8(k1 + 16 + kh * 8);
            b11.h[0] = load_h8(k1 + 32 + kh * 8); b11.h[1] = load_h8(k1 + 48 + kh * 8);
        }
        #pragma unroll
        for (int sub = 0; sub < 4; ++sub) {
            const _Float16* vp = Vp + (size_t)(sub * 16 + n) * SEQ + kbase;
            bv[sub].h[0] = load_h8(vp + kh * 8);
            bv[sub].h[1] = load_h8(vp + 16 + kh * 8);
        }

        // --- scores (only needs the first 8 loads) ---
        v8f s0 = {}, s1 = {};
        s0 = wmma16x16x32(aq0, b00, s0);
        s0 = wmma16x16x32(aq1, b01, s0);
        s1 = wmma16x16x32(aq0, b10, s1);
        s1 = wmma16x16x32(aq1, b11, s1);

        // --- causal mask + online softmax (V loads still in flight) ---
        float p0[8], p1[8], alpha[8];
        #pragma unroll
        for (int v = 0; v < 8; ++v) {
            const int q = q0 + v + 8 * kh;
            float x0 = (kbase + n      <= q) ? s0[v] : -__builtin_inff();
            float x1 = (kbase + 16 + n <= q) ? s1[v] : -__builtin_inff();
            float mx = fmaxf(x0, x1);
            #pragma unroll
            for (int off = 8; off >= 1; off >>= 1)
                mx = fmaxf(mx, __shfl_xor(mx, off, 32));   // reduce over 16-lane half
            const float mnew = fmaxf(mrun[v], mx);
            const float a    = __expf(mrun[v] - mnew);     // 0 on first tile
            mrun[v] = mnew;
            p0[v] = __expf(x0 - mnew);
            p1[v] = __expf(x1 - mnew);
            float rs = p0[v] + p1[v];
            #pragma unroll
            for (int off = 8; off >= 1; off >>= 1)
                rs += __shfl_xor(rs, off, 32);
            lrun[v]  = lrun[v] * a + rs;
            alpha[v] = a;
        }

        #pragma unroll
        for (int sub = 0; sub < 4; ++sub)
            #pragma unroll
            for (int v = 0; v < 8; ++v) o[sub][v] *= alpha[v];

        // --- P: C-layout -> LDS -> A-layout (16x32 f16) ---
        #pragma unroll
        for (int v = 0; v < 8; ++v) {
            pt[v + 8 * kh][n]      = (_Float16)p0[v];
            pt[v + 8 * kh][n + 16] = (_Float16)p1[v];
        }
        __syncthreads();   // single-wave block: barrier -> S_NOP; keeps DS ordering
        AFrag ap;
        ap.h[0] = *reinterpret_cast<const h8*>(&pt[n][kh * 8]);
        ap.h[1] = *reinterpret_cast<const h8*>(&pt[n][16 + kh * 8]);

        // --- O += P(16x32) x V(32x64) ---
        #pragma unroll
        for (int sub = 0; sub < 4; ++sub)
            o[sub] = wmma16x16x32(ap, bv[sub], o[sub]);
        __syncthreads();
    }

    // normalize rows and store f16 attention output [B,T,768]
    const int bb = bh / NHEADS, h = bh % NHEADS;
    #pragma unroll
    for (int v = 0; v < 8; ++v) {
        const float inv = 1.0f / lrun[v];
        const int t = q0 + v + 8 * kh;
        _Float16* orow = attnb + ((size_t)(bb * SEQ + t)) * DMODEL + h * DHEAD;
        #pragma unroll
        for (int sub = 0; sub < 4; ++sub)
            orow[sub * 16 + n] = (_Float16)(o[sub][v] * inv);
    }
}

// ---------------------------------------------------------------------------
// Output projection: out[row][e] = sum_k attn[row][k] * w_out[e][k] (f32 out)
// 32x64 tile per wave, double-buffered K-loop.
// ---------------------------------------------------------------------------
__global__ __launch_bounds__(128) void out_gemm(
    const _Float16* __restrict__ ab, const _Float16* __restrict__ wo,
    float* __restrict__ out)
{
    const int wave = threadIdx.x >> 5;
    const int lane = threadIdx.x & 31;
    const int task = blockIdx.x * 4 + wave;   // 256 row-tiles(32) * 12 col-tiles(64)
    const int rt = task / 12;
    const int ct = task % 12;
    const int m  = lane & 15;
    const int kh = lane >> 4;

    const _Float16* arow0 = ab + (size_t)(rt * 32 + m) * DMODEL;
    const _Float16* arow1 = arow0 + (size_t)16 * DMODEL;
    const _Float16* wbase = wo + (size_t)(ct * 64 + m) * DMODEL;

    AFrag A[2][2], B[2][4];
    v8f acc[8] = {};

    auto loadA = [&](AFrag* dst, int kt) {
        dst[0].h[0] = load_h8(arow0 + kt + kh * 8);
        dst[0].h[1] = load_h8(arow0 + kt + 16 + kh * 8);
        dst[1].h[0] = load_h8(arow1 + kt + kh * 8);
        dst[1].h[1] = load_h8(arow1 + kt + 16 + kh * 8);
    };
    auto loadB = [&](AFrag* dst, int kt) {
        #pragma unroll
        for (int sub = 0; sub < 4; ++sub) {
            const _Float16* wrow = wbase + (size_t)sub * 16 * DMODEL + kt;
            dst[sub].h[0] = load_h8(wrow + kh * 8);
            dst[sub].h[1] = load_h8(wrow + 16 + kh * 8);
        }
    };

    loadA(A[0], 0);
    loadB(B[0], 0);
    int pb = 0;
    for (int kt = 0; kt < DMODEL; kt += 32, pb ^= 1) {
        if (kt + 32 < DMODEL) {
            loadA(A[pb ^ 1], kt + 32);
            loadB(B[pb ^ 1], kt + 32);
        }
        #pragma unroll
        for (int sub = 0; sub < 4; ++sub)
            acc[sub] = wmma16x16x32(A[pb][0], B[pb][sub], acc[sub]);
        #pragma unroll
        for (int sub = 0; sub < 4; ++sub)
            acc[4 + sub] = wmma16x16x32(A[pb][1], B[pb][sub], acc[4 + sub]);
    }

    #pragma unroll
    for (int rb = 0; rb < 2; ++rb) {
        #pragma unroll
        for (int sub = 0; sub < 4; ++sub) {
            const int col = ct * 64 + sub * 16 + m;
            #pragma unroll
            for (int v = 0; v < 8; ++v) {
                const int row = rt * 32 + rb * 16 + v + 8 * kh;
                out[(size_t)row * DMODEL + col] = acc[rb * 4 + sub][v];
            }
        }
    }
}

// ---------------------------------------------------------------------------
extern "C" void kernel_launch(void* const* d_in, const int* in_sizes, int n_in,
                              void* d_out, int out_size, void* d_ws, size_t ws_size,
                              hipStream_t stream) {
    const float* x    = (const float*)d_in[0];
    const float* wqkv = (const float*)d_in[1];
    const float* wout = (const float*)d_in[2];

    size_t off = 0;
    auto carve = [&](size_t bytes) {
        void* p = (char*)d_ws + off;
        off += (bytes + 255) & ~(size_t)255;
        return p;
    };
    const size_t qkv_elems = (size_t)BATCH * NHEADS * SEQ * DHEAD;  // 6.29M
    _Float16* xb  = (_Float16*)carve((size_t)ROWS * DMODEL * 2);    // reused as attnb
    _Float16* wqb = (_Float16*)carve((size_t)3 * DMODEL * DMODEL * 2);
    _Float16* wob = (_Float16*)carve((size_t)DMODEL * DMODEL * 2);
    _Float16* Qh  = (_Float16*)carve(qkv_elems * 2);
    _Float16* Kh  = (_Float16*)carve(qkv_elems * 2);
    _Float16* Vt  = (_Float16*)carve(qkv_elems * 2);
    (void)ws_size; (void)in_sizes; (void)n_in; (void)out_size;

    const int n1 = ROWS * DMODEL;
    const int n2 = 3 * DMODEL * DMODEL;
    const int n3 = DMODEL * DMODEL;
    convert_f32_f16<<<(n1 + 255) / 256, 256, 0, stream>>>(x, xb, n1);
    convert_f32_f16<<<(n2 + 255) / 256, 256, 0, stream>>>(wqkv, wqb, n2);
    convert_f32_f16<<<(n3 + 255) / 256, 256, 0, stream>>>(wout, wob, n3);

    // 256 row-tiles * 36 col-tiles, 4 waves/block
    qkv_gemm<<<(256 * 36) / 4, 128, 0, stream>>>(xb, wqb, Qh, Kh, Vt);

    // B*H*(T/16) = 6144 single-wave blocks; writes attn output into xb (reuse)
    attn_kernel<<<BATCH * NHEADS * (SEQ / 16), 32, 0, stream>>>(Qh, Kh, Vt, xb);

    // 256 row-tiles * 12 col-tiles, 4 waves/block
    out_gemm<<<(256 * 12) / 4, 128, 0, stream>>>(xb, wob, (float*)d_out);
}
